// CustomRNN_53180285059162
// MI455X (gfx1250) — matrix operand order repaired and checked by reading, not compiled, not run on hardware
//
#include <hip/hip_runtime.h>
#include <math.h>

typedef __attribute__((ext_vector_type(16))) _Float16 v16h;
typedef __attribute__((ext_vector_type(8)))  _Float16 v8h;
typedef __attribute__((ext_vector_type(8)))  float    v8f;
typedef __attribute__((ext_vector_type(4)))  unsigned int u32x4;
typedef __attribute__((ext_vector_type(8)))  unsigned int u32x8;

static constexpr int Bv = 64, Sv = 512, Iv = 256, Hv = 512, Ov = 256, Lv = 2;

__device__ __forceinline__ v16h cat8(v8h lo, v8h hi) {
  v16h r;
#pragma unroll
  for (int i = 0; i < 8; ++i) { r[i] = lo[i]; r[i + 8] = hi[i]; }
  return r;
}

// A fragment: 16x32 f16 tile, row-major source, leading dim ld, k offset k0.
// Lane L holds row (L&15), halves 0-7 = K[k0 + (L>>4)*8 ..], halves 8-15 = +16.
__device__ __forceinline__ v16h load_a16(const _Float16* tile, int ld, int k0) {
  const int lane = threadIdx.x & 31;
  const _Float16* p = tile + (size_t)(lane & 15) * ld + k0 + ((lane >> 4) << 3);
  return cat8(*(const v8h*)p, *(const v8h*)(p + 16));
}

// B fragment for D = A * W^T: W is [N x K] row-major; tile points at row n0.
// Lane L holds column n=(L&15), halves 0-7 = K[k0 + (L>>4)*8 ..], halves 8-15 = +16.
__device__ __forceinline__ v16h load_b16(const _Float16* tile, int ld, int k0) {
  const int lane = threadIdx.x & 31;
  const _Float16* p = tile + (size_t)(lane & 15) * ld + k0 + ((lane >> 4) << 3);
  return cat8(*(const v8h*)p, *(const v8h*)(p + 16));
}

__device__ __forceinline__ v8f wmma16(v16h a, v16h b, v8f c) {
  return __builtin_amdgcn_wmma_f32_16x16x32_f16(false, a, false, b, (short)0, c,
                                                false, false);
}

// TDM: DMA a 2D tile (tile_dim1=64 rows x tile_dim0=512 f16, row stride S*H
// elements) from global `gptr` into LDS byte offset `lds_byte_off`.
// D# layout per CDNA5 ISA ch.8; groups 2/3 zero => tile dims 2+ unused.
__device__ __forceinline__ void tdm_load_xe_tile(const _Float16* gptr,
                                                 unsigned int lds_byte_off) {
  unsigned long long ga = (unsigned long long)(const void*)gptr;
  u32x4 g0;
  g0[0] = 1u;                                    // count=1, user descriptor
  g0[1] = lds_byte_off;                          // lds_addr
  g0[2] = (unsigned int)ga;                      // global_addr[31:0]
  g0[3] = (unsigned int)(ga >> 32) | (2u << 30); // global_addr[56:32] | type=2
  u32x8 g1;
  g1[0] = 1u << 16;                              // data_size=1 (2 bytes)
  g1[1] = (unsigned int)Hv << 16;                // tensor_dim0[15:0]=512
  g1[2] = (unsigned int)Bv << 16;                // tensor_dim1[15:0]=64
  g1[3] = (unsigned int)Hv << 16;                // tile_dim0=512
  g1[4] = (unsigned int)Bv;                      // tile_dim1=64, tile_dim2=0
  g1[5] = (unsigned int)(Sv * Hv);               // tensor_dim0_stride=S*H
  g1[6] = 0u;
  g1[7] = 0u;
  u32x4 g2 = {0u, 0u, 0u, 0u};
  u32x4 g3 = {0u, 0u, 0u, 0u};
  asm volatile("tensor_load_to_lds %0, %1, %2, %3"
               :: "s"(g0), "s"(g1), "s"(g2), "s"(g3)
               : "memory");
}

// ---------------------------------------------------------------- converts
__global__ void f32_to_f16_kernel(const float* __restrict__ src,
                                  _Float16* __restrict__ dst, int n) {
  int i = blockIdx.x * blockDim.x + threadIdx.x;
  if (i < n) dst[i] = (_Float16)src[i];
}

// ------------------------------------------------- input projection (WMMA)
// xe[B*S, H] = x16[B*S, I] @ inW16[H, I]^T + inB
__global__ __launch_bounds__(256)
void input_proj_kernel(const _Float16* __restrict__ x16,
                       const _Float16* __restrict__ w16,
                       const float* __restrict__ bias,
                       _Float16* __restrict__ xe) {
  const int wave = blockIdx.x * 8 + (threadIdx.x >> 5);
  const int lane = threadIdx.x & 31;
  const int col  = lane & 15;
  const int sel  = lane >> 4;
  const int mt = wave >> 5;  // 0..2047 (rows of B*S)
  const int nt = wave & 31;  // 0..31   (cols of H)

  v8f acc;
  const float b0 = bias[nt * 16 + col];
#pragma unroll
  for (int r = 0; r < 8; ++r) acc[r] = b0;

  const _Float16* A0 = x16 + (size_t)mt * 16 * Iv;
  const _Float16* B0 = w16 + (size_t)nt * 16 * Iv;
#pragma unroll
  for (int k0 = 0; k0 < Iv; k0 += 32) {
    v16h a  = load_a16(A0, Iv, k0);
    v16h bb = load_b16(B0, Iv, k0);
    acc = wmma16(a, bb, acc);
  }
  _Float16* dst = xe + (size_t)mt * 16 * Hv + nt * 16;
#pragma unroll
  for (int r = 0; r < 8; ++r)
    dst[(size_t)(r + sel * 8) * Hv + col] = (_Float16)acc[r];
}

// ---------------------------------------------- persistent recurrence scan
// One workgroup (32 wave32s) owns the scan. LDS: h0 | h1 | xe ping | xe pong
// (64 KB each = 256 KB of the WGP's 320 KB). xe tiles arrive via TDM.
__global__ __launch_bounds__(1024)
void rnn_scan_kernel(const _Float16* __restrict__ xe,
                     const _Float16* __restrict__ i2h,
                     const _Float16* __restrict__ h2h,
                     const float* __restrict__ h2h_b,
                     const _Float16* __restrict__ outW,
                     const float* __restrict__ out_b,
                     float* __restrict__ out) {
  extern __shared__ _Float16 lds[];
  _Float16* hbuf0 = lds;                    // LDS bytes [0, 64K)
  _Float16* hbuf1 = lds + Bv * Hv;          // [64K, 128K)
  // xe ping/pong buffers live at slots 2 and 3; compute pointers on the fly
  // (no arrays of LDS-derived pointers: that forces an unlinkable
  //  addrspacecast static initializer).

  const int tid  = threadIdx.x;
  const int wave = tid >> 5;
  const int lane = tid & 31;
  const int col  = lane & 15;
  const int sel  = lane >> 4;

  for (int i = tid; i < 2 * Bv * Hv; i += 1024) lds[i] = (_Float16)0.0f;

  // prefetch xe tile for t=0 (wave 0 only; TDM ignores EXEC, gate per wave)
  if (wave == 0) tdm_load_xe_tile(xe, 2u * Bv * Hv * 2u);

  __syncthreads();

  const int nt = wave;  // this wave's output column tile (0..31)
  const float bias0 = h2h_b[0 * Hv + nt * 16 + col];
  const float bias1 = h2h_b[1 * Hv + nt * 16 + col];

  for (int t = 0; t < Sv; ++t) {
    const int cur = t & 1;
    if (wave == 0) {
      if (t + 1 < Sv) {  // prefetch t+1, then wait for t (in-order per wave)
        tdm_load_xe_tile(xe + (size_t)(t + 1) * Hv,
                         (2u + (unsigned)((t + 1) & 1)) * (unsigned)(Bv * Hv) * 2u);
        __builtin_amdgcn_s_wait_tensorcnt(1);
      } else {
        __builtin_amdgcn_s_wait_tensorcnt(0);
      }
    }
    __syncthreads();  // xe_t visible in LDS to all waves

    const _Float16* xcur = lds + (size_t)(2 + cur) * Bv * Hv;

#pragma unroll
    for (int l = 0; l < Lv; ++l) {
      const _Float16* Wx = i2h + (size_t)l * Hv * Hv + (size_t)nt * 16 * Hv;
      const _Float16* Wh = h2h + (size_t)l * Hv * Hv + (size_t)nt * 16 * Hv;
      const _Float16* in = (l == 0) ? xcur : hbuf0;  // layer input
      _Float16* hl = (l == 0) ? hbuf0 : hbuf1;       // recurrent state
      const float bias = (l == 0) ? bias0 : bias1;

      v8f acc[4];
#pragma unroll
      for (int m = 0; m < 4; ++m)
#pragma unroll
        for (int r = 0; r < 8; ++r) acc[m][r] = bias;

      for (int k0 = 0; k0 < Hv; k0 += 32) {
        v16h bx = load_b16(Wx, Hv, k0);  // reused across 4 row tiles
        v16h bh = load_b16(Wh, Hv, k0);
#pragma unroll
        for (int m = 0; m < 4; ++m) {
          v16h ax = load_a16(in + (size_t)m * 16 * Hv, Hv, k0);
          acc[m] = wmma16(ax, bx, acc[m]);
          v16h ah = load_a16(hl + (size_t)m * 16 * Hv, Hv, k0);  // h_prev[l]
          acc[m] = wmma16(ah, bh, acc[m]);
        }
      }
      __syncthreads();  // everyone done reading h_prev[l]
#pragma unroll
      for (int m = 0; m < 4; ++m) {
        _Float16* dst = hl + (size_t)m * 16 * Hv + nt * 16;
#pragma unroll
        for (int r = 0; r < 8; ++r)
          dst[(size_t)(r + sel * 8) * Hv + col] = (_Float16)tanhf(acc[m][r]);
      }
      __syncthreads();  // h_new[l] visible to all waves
    }
  }

  // hT (f32) -> out[B*O ..]  (lds layout == [L,B,H] flat)
  for (int i = tid; i < Lv * Bv * Hv; i += 1024)
    out[Bv * Ov + i] = (float)lds[i];

  // output head: out[b,o] = h1[b,:] . outW[o,:] + out_b[o]   (4x16 tiles)
  {
    const int on    = wave & 15;
    const int mbase = (wave >> 4) * 2;
    const _Float16* Wo = outW + (size_t)on * 16 * Hv;
    const float bo = out_b[on * 16 + col];
#pragma unroll
    for (int mm = 0; mm < 2; ++mm) {
      const int m = mbase + mm;
      v8f acc;
#pragma unroll
      for (int r = 0; r < 8; ++r) acc[r] = bo;
#pragma unroll 4
      for (int k0 = 0; k0 < Hv; k0 += 32) {
        v16h a = load_a16(hbuf1 + (size_t)m * 16 * Hv, Hv, k0);
        v16h b = load_b16(Wo, Hv, k0);
        acc = wmma16(a, b, acc);
      }
#pragma unroll
      for (int r = 0; r < 8; ++r) {
        const int row = m * 16 + r + sel * 8;
        out[(size_t)row * Ov + on * 16 + col] = acc[r];
      }
    }
  }
}

// ---------------------------------------------------------------- launcher
extern "C" void kernel_launch(void* const* d_in, const int* in_sizes, int n_in,
                              void* d_out, int out_size, void* d_ws, size_t ws_size,
                              hipStream_t stream) {
  (void)in_sizes; (void)n_in; (void)out_size; (void)ws_size;
  const float* x    = (const float*)d_in[0];
  const float* inW  = (const float*)d_in[1];
  const float* inB  = (const float*)d_in[2];
  const float* i2hW = (const float*)d_in[3];
  const float* h2hW = (const float*)d_in[4];
  const float* h2hB = (const float*)d_in[5];
  const float* outW = (const float*)d_in[6];
  const float* outB = (const float*)d_in[7];
  float* out = (float*)d_out;

  _Float16* ws = (_Float16*)d_ws;
  size_t off = 0;
  _Float16* x16    = ws + off; off += (size_t)Bv * Sv * Iv;  // 16 MB
  _Float16* xe16   = ws + off; off += (size_t)Bv * Sv * Hv;  // 32 MB
  _Float16* inW16  = ws + off; off += (size_t)Hv * Iv;
  _Float16* i2h16  = ws + off; off += (size_t)Lv * Hv * Hv;
  _Float16* h2h16  = ws + off; off += (size_t)Lv * Hv * Hv;
  _Float16* outW16 = ws + off; off += (size_t)Ov * Hv;

  auto cvt = [&](const float* s, _Float16* d, int n) {
    f32_to_f16_kernel<<<(n + 255) / 256, 256, 0, stream>>>(s, d, n);
  };
  cvt(x,    x16,    Bv * Sv * Iv);
  cvt(inW,  inW16,  Hv * Iv);
  cvt(i2hW, i2h16,  Lv * Hv * Hv);
  cvt(h2hW, h2h16,  Lv * Hv * Hv);
  cvt(outW, outW16, Ov * Hv);

  // 2048 row-tiles x 32 col-tiles, 8 waves per block
  input_proj_kernel<<<(Bv * Sv / 16) * (Hv / 16) / 8, 256, 0, stream>>>(
      x16, inW16, inB, xe16);

  rnn_scan_kernel<<<1, 1024, 4 * Bv * Hv * (int)sizeof(_Float16), stream>>>(
      xe16, i2h16, h2h16, h2hB, outW16, outB, out);
}
